// StandardEuclidean_45234595561962
// MI455X (gfx1250) — matrix operand order repaired and checked by reading, not compiled, not run on hardware
//
#include <hip/hip_runtime.h>
#include <hip/hip_bf16.h>

#define B_ 32
#define N_ 8192
#define D_ 64
#define H_ 128
#define S_ 16

typedef __bf16 bf16;
typedef bf16  bf16x16 __attribute__((ext_vector_type(16)));
typedef float f32x8   __attribute__((ext_vector_type(8)));

__device__ __forceinline__ f32x8 wmma_bf16(bf16x16 a, bf16x16 b, f32x8 c) {
  // D = A(16x32) * B(32x16) + C, f32 accum
  return __builtin_amdgcn_wmma_f32_16x16x32_bf16(false, a, false, b, (short)0, c,
                                                 false, false);
}

// A-fragment: 16x32 bf16 tile, row-major source, stride lda (vectorized).
// Lane l: row = l&15, kb = (l>>4)*16; elements 0..15 -> K kb..kb+15.
__device__ __forceinline__ bf16x16 load_a_frag(const bf16* src, int lda, int lane) {
  const bf16* p = src + (size_t)(lane & 15) * lda + ((lane >> 4) << 4);
  union { bf16x16 v; uint4 q[2]; } u;
  u.q[0] = *(const uint4*)p;
  u.q[1] = *(const uint4*)(p + 8);
  return u.v;
}

// 16 contiguous bf16 (32 bytes) -> fragment register file.
__device__ __forceinline__ bf16x16 load_contig16(const bf16* p) {
  union { bf16x16 v; uint4 q[2]; } u;
  u.q[0] = *(const uint4*)p;
  u.q[1] = *(const uint4*)(p + 8);
  return u.v;
}

__device__ __forceinline__ float gelu_exact(float v) {
  return v * 0.5f * (1.0f + erff(v * 0.70710678118654752f));
}

// ---- CDNA5 async copy: global -> LDS, 16B per lane, tracked by ASYNCcnt ----
__device__ __forceinline__ void async_load_b128(unsigned lds_off, const void* gaddr) {
  asm volatile("global_load_async_to_lds_b128 %0, %1, off"
               :
               : "v"(lds_off), "v"(gaddr)
               : "memory");
}
__device__ __forceinline__ void wait_async_le16() {
#if __has_builtin(__builtin_amdgcn_s_wait_asynccnt)
  __builtin_amdgcn_s_wait_asynccnt(16);
#else
  asm volatile("s_wait_asynccnt 0x10" ::: "memory");
#endif
  asm volatile("" ::: "memory");
}
__device__ __forceinline__ void wait_async_le0() {
#if __has_builtin(__builtin_amdgcn_s_wait_asynccnt)
  __builtin_amdgcn_s_wait_asynccnt(0);
#else
  asm volatile("s_wait_asynccnt 0x0" ::: "memory");
#endif
  asm volatile("" ::: "memory");
}

// ---------------------------------------------------------------------------
// Kernel A: feats = LN2(GELU(LN1(x@w1+b1))@w2+b2)  [bf16 out, row-major and
// transposed], f2 = sum feats^2.  grid = B*N/128, 256 threads (8 waves).
// ---------------------------------------------------------------------------
__global__ __launch_bounds__(256) void feats_kernel(
    const float* __restrict__ x, const float* __restrict__ w1,
    const float* __restrict__ b1, const float* __restrict__ g1,
    const float* __restrict__ bb1, const float* __restrict__ w2,
    const float* __restrict__ b2, const float* __restrict__ g2,
    const float* __restrict__ bb2, bf16* __restrict__ feats,
    bf16* __restrict__ featsT, float* __restrict__ f2out) {
  // fragment-packed weights: [frag][lane][16 bf16], each lane slice 32B.
  __shared__ __align__(16) bf16 s_w1p[16 * 32 * 16];  // 2 kc * 8 ct
  __shared__ __align__(16) bf16 s_w2p[32 * 32 * 16];  // 4 kc * 8 ct
  __shared__ __align__(16) bf16 s_h[8][16 * H_];

  const int tid = threadIdx.x;
  const int lane = tid & 31, wave = tid >> 5;
  const int bidx = blockIdx.x / (N_ / 128);
  const int nblk = blockIdx.x % (N_ / 128);
  const int row0 = nblk * 128 + wave * 16;
  const int colq = lane & 15;
  const int hi = lane >> 4;

  // ---- pack weights into fragment-ready LDS layout (2x ds_load_b128 / frag) ----
  for (int pi = wave; pi < 16 * 16; pi += 8) {  // (frag, i) pairs for w1
    int frag = pi >> 4, i = pi & 15;
    int kc = frag >> 3, ct = frag & 7;
    float v = w1[(size_t)(kc * 32 + ((lane >> 4) << 4) + i) * H_ + ct * 16 + (lane & 15)];
    s_w1p[((size_t)frag * 32 + lane) * 16 + i] = (bf16)v;
  }
  for (int pi = wave; pi < 32 * 16; pi += 8) {  // (frag, i) pairs for w2
    int frag = pi >> 4, i = pi & 15;
    int kc = frag >> 3, ct = frag & 7;
    float v = w2[(size_t)(kc * 32 + ((lane >> 4) << 4) + i) * H_ + ct * 16 + (lane & 15)];
    s_w2p[((size_t)frag * 32 + lane) * 16 + i] = (bf16)v;
  }
  __syncthreads();

  // ---- A-fragments of x (16 rows x 64 K, two K-chunks), f32 -> bf16 ----
  bf16x16 ax[2];
  {
    const float* xr = x + ((size_t)bidx * N_ + row0 + colq) * D_;
#pragma unroll
    for (int kc = 0; kc < 2; ++kc) {
      const float* p = xr + kc * 32 + (hi << 4);
      float4 fa = *(const float4*)(p);
      float4 fb4 = *(const float4*)(p + 4);
      float4 fc = *(const float4*)(p + 8);
      float4 fd = *(const float4*)(p + 12);
      bf16x16 a;
      a[0] = (bf16)fa.x;  a[1] = (bf16)fa.y;  a[2] = (bf16)fa.z;  a[3] = (bf16)fa.w;
      a[4] = (bf16)fb4.x; a[5] = (bf16)fb4.y; a[6] = (bf16)fb4.z; a[7] = (bf16)fb4.w;
      a[8] = (bf16)fc.x;  a[9] = (bf16)fc.y;  a[10] = (bf16)fc.z; a[11] = (bf16)fc.w;
      a[12] = (bf16)fd.x; a[13] = (bf16)fd.y; a[14] = (bf16)fd.z; a[15] = (bf16)fd.w;
      ax[kc] = a;
    }
  }

  // ---- GEMM1: t = x@w1 + b1 ----
  f32x8 t[8];
#pragma unroll
  for (int ct = 0; ct < 8; ++ct) {
    float bv = b1[ct * 16 + colq];
    f32x8 c = {bv, bv, bv, bv, bv, bv, bv, bv};
    c = wmma_bf16(ax[0], load_contig16(&s_w1p[((size_t)(0 * 8 + ct) * 32 + lane) * 16]), c);
    c = wmma_bf16(ax[1], load_contig16(&s_w1p[((size_t)(1 * 8 + ct) * 32 + lane) * 16]), c);
    t[ct] = c;
  }

  // ---- LN1 + GELU (row stats via half-wave shuffles) ----
  float mean[8], rstd[8];
#pragma unroll
  for (int r = 0; r < 8; ++r) {
    float s = 0.f, sq = 0.f;
#pragma unroll
    for (int ct = 0; ct < 8; ++ct) {
      float v = t[ct][r];
      s += v;
      sq += v * v;
    }
#pragma unroll
    for (int m = 1; m < 16; m <<= 1) {
      s += __shfl_xor(s, m, 32);
      sq += __shfl_xor(sq, m, 32);
    }
    float mu = s * (1.0f / H_);
    mean[r] = mu;
    rstd[r] = rsqrtf(sq * (1.0f / H_) - mu * mu + 1e-5f);
  }
#pragma unroll
  for (int ct = 0; ct < 8; ++ct) {
    int col = ct * 16 + colq;
    float g = g1[col], bb = bb1[col];
#pragma unroll
    for (int r = 0; r < 8; ++r) {
      float v = (t[ct][r] - mean[r]) * rstd[r] * g + bb;
      t[ct][r] = gelu_exact(v);
    }
  }

  // ---- stage h (bf16) in LDS for second GEMM A-fragments ----
#pragma unroll
  for (int ct = 0; ct < 8; ++ct)
#pragma unroll
    for (int r = 0; r < 8; ++r)
      s_h[wave][(size_t)(r + (hi << 3)) * H_ + ct * 16 + colq] = (bf16)t[ct][r];
  __syncthreads();

  bf16x16 ah[4];
#pragma unroll
  for (int kc = 0; kc < 4; ++kc)
    ah[kc] = load_a_frag(&s_h[wave][kc * 32], H_, lane);

  // ---- GEMM2: t2 = h@w2 + b2 ----
  f32x8 t2[8];
#pragma unroll
  for (int ct = 0; ct < 8; ++ct) {
    float bv = b2[ct * 16 + colq];
    f32x8 c = {bv, bv, bv, bv, bv, bv, bv, bv};
#pragma unroll
    for (int kc = 0; kc < 4; ++kc)
      c = wmma_bf16(ah[kc], load_contig16(&s_w2p[((size_t)(kc * 8 + ct) * 32 + lane) * 16]), c);
    t2[ct] = c;
  }

  // ---- LN2 + f2 ----
  float f2p[8];
#pragma unroll
  for (int r = 0; r < 8; ++r) {
    float s = 0.f, sq = 0.f;
#pragma unroll
    for (int ct = 0; ct < 8; ++ct) {
      float v = t2[ct][r];
      s += v;
      sq += v * v;
    }
#pragma unroll
    for (int m = 1; m < 16; m <<= 1) {
      s += __shfl_xor(s, m, 32);
      sq += __shfl_xor(sq, m, 32);
    }
    float mu = s * (1.0f / H_);
    mean[r] = mu;
    rstd[r] = rsqrtf(sq * (1.0f / H_) - mu * mu + 1e-5f);
    f2p[r] = 0.f;
  }
#pragma unroll
  for (int ct = 0; ct < 8; ++ct) {
    int col = ct * 16 + colq;
    float g = g2[col], bb = bb2[col];
#pragma unroll
    for (int r = 0; r < 8; ++r) {
      float v = (t2[ct][r] - mean[r]) * rstd[r] * g + bb;
      t2[ct][r] = v;
      f2p[r] += v * v;
    }
  }
#pragma unroll
  for (int r = 0; r < 8; ++r) {
#pragma unroll
    for (int m = 1; m < 16; m <<= 1) f2p[r] += __shfl_xor(f2p[r], m, 32);
  }

  // restage feats (bf16) in LDS; whole block then owns a 128x128 tile
#pragma unroll
  for (int ct = 0; ct < 8; ++ct)
#pragma unroll
    for (int r = 0; r < 8; ++r)
      s_h[wave][(size_t)(r + (hi << 3)) * H_ + ct * 16 + colq] = (bf16)t2[ct][r];
  __syncthreads();

  if (colq == 0) {
#pragma unroll
    for (int r = 0; r < 8; ++r)
      f2out[(size_t)bidx * N_ + row0 + (hi << 3) + r] = f2p[r];
  }
  // row-major feats store, coalesced 16B
  {
    const uint4* src = (const uint4*)&s_h[wave][0];  // 256 uint4 per wave
    uint4* dst = (uint4*)(feats + ((size_t)bidx * N_ + row0) * H_);
#pragma unroll
    for (int j = 0; j < 8; ++j) dst[lane + 32 * j] = src[lane + 32 * j];
  }
  // transposed feats store: featsT[b][h][nblk*128 .. +128), coalesced 16B rows
  {
    bf16* ftb = featsT + (size_t)bidx * H_ * N_ + nblk * 128;
#pragma unroll
    for (int j = 0; j < 8; ++j) {
      int u = tid + 256 * j;  // uint4 id 0..2047 (128 h x 16 chunks)
      int h = u >> 4;
      int nq = u & 15;        // 8-element chunk in n
      int w = nq >> 1;
      int r0 = (nq & 1) * 8;
      union { uint4 q; bf16 e[8]; } pk;
#pragma unroll
      for (int e = 0; e < 8; ++e) pk.e[e] = s_h[w][(size_t)(r0 + e) * H_ + h];
      *(uint4*)(ftb + (size_t)h * N_ + nq * 8) = pk.q;
    }
  }
}

// ---------------------------------------------------------------------------
// Kernel B: slot-attention iterations. grid = B blocks, 256 threads (8 waves).
// feats tiles stream through LDS via async copies, double-buffered.
// ---------------------------------------------------------------------------
__global__ __launch_bounds__(256) void slots_kernel(
    const bf16* __restrict__ feats, const bf16* __restrict__ featsT,
    const float* __restrict__ f2g, const float* __restrict__ slot_mu,
    const float* __restrict__ gru_wi, const float* __restrict__ gru_wh,
    const float* __restrict__ gru_bi, const float* __restrict__ gru_bh,
    const float* __restrict__ mlp_w1, const float* __restrict__ mlp_b1,
    const float* __restrict__ mlp_w2, const float* __restrict__ mlp_b2,
    const float* __restrict__ norm_g, const float* __restrict__ norm_b,
    float* __restrict__ out_slots, float* __restrict__ out_attn) {
  __shared__ __align__(16) bf16 s_ftile[2][256 * H_];  // 2 x 64 KB double buffer
  __shared__ float s_slots[S_][H_];
  __shared__ __align__(16) bf16 s_slotb[S_][H_];
  __shared__ float s_upd[S_][H_];
  __shared__ float s_new[S_][H_];
  __shared__ float s_buf[S_][H_];
  __shared__ float s_s2[S_];
  __shared__ float s_asum[S_];
  __shared__ __align__(16) bf16 s_stage[8][S_][32];

  const int tid = threadIdx.x, lane = tid & 31, wave = tid >> 5;
  const int bidx = blockIdx.x;
  const int colq = lane & 15, hi = lane >> 4;
  const int sg = tid >> 4;  // slot group 0..15 (16 threads each)
  const int jj = tid & 15;

  for (int i = tid; i < S_ * H_; i += 256) (&s_slots[0][0])[i] = slot_mu[i];
  __syncthreads();

  const bf16* fb = feats + (size_t)bidx * N_ * H_;
  const bf16* ftb = featsT + (size_t)bidx * H_ * N_;
  const float* f2b = f2g + (size_t)bidx * N_;

  // per-wave LDS byte offsets of this wave's 32x128 bf16 tile region
  const unsigned ldsw0 =
      (unsigned)(size_t)(void*)&s_ftile[0][0] + (unsigned)wave * 8192u;
  const unsigned ldsw1 =
      (unsigned)(size_t)(void*)&s_ftile[1][0] + (unsigned)wave * 8192u;

  for (int iter = 0; iter < 3; ++iter) {
    // ---- s2, bf16 slots, zero accumulators ----
    {
      float p = 0.f;
#pragma unroll
      for (int k = 0; k < 8; ++k) {
        float v = s_slots[sg][jj + 16 * k];
        s_slotb[sg][jj + 16 * k] = (bf16)v;
        p += v * v;
      }
#pragma unroll
      for (int m = 1; m < 16; m <<= 1) p += __shfl_xor(p, m, 32);
      if (jj == 0) s_s2[sg] = p;
    }
    for (int i = tid; i < S_ * H_; i += 256) (&s_upd[0][0])[i] = 0.f;
    if (tid < S_) s_asum[tid] = 0.f;
    __syncthreads();

    // A-fragments of slots (K = H = 128, 4 chunks); reused over all of N
    bf16x16 as[4];
#pragma unroll
    for (int kc = 0; kc < 4; ++kc) as[kc] = load_a_frag(&s_slotb[0][kc * 32], H_, lane);

    float apart[8];
    f32x8 cupd[8];
#pragma unroll
    for (int r = 0; r < 8; ++r) apart[r] = 0.f;
#pragma unroll
    for (int ht = 0; ht < 8; ++ht) {
      f32x8 z = {0.f, 0.f, 0.f, 0.f, 0.f, 0.f, 0.f, 0.f};
      cupd[ht] = z;
    }

    // ---- prologue: async-load tile 0 into buffer 0 (16x 512B per wave) ----
    {
      const char* g = (const char*)(fb + (size_t)(wave * 32) * H_);
#pragma unroll
      for (int j = 0; j < 16; ++j)
        async_load_b128(ldsw0 + (unsigned)(j * 32 + lane) * 16u,
                        g + (size_t)(j * 32 + lane) * 16);
    }

    for (int nt = 0; nt < N_; nt += 256) {
      const int n0 = nt + wave * 32;
      const int buf = (nt >> 8) & 1;
      if (nt + 256 < N_) {
        // issue async copies for next tile into the other buffer, then wait
        // until the previous 16 (current tile) have landed: ASYNCcnt <= 16.
        const unsigned ldsn = (buf ? ldsw0 : ldsw1);
        const char* g = (const char*)(fb + (size_t)(n0 + 256) * H_);
#pragma unroll
        for (int j = 0; j < 16; ++j)
          async_load_b128(ldsn + (unsigned)(j * 32 + lane) * 16u,
                          g + (size_t)(j * 32 + lane) * 16);
        wait_async_le16();
      } else {
        wait_async_le0();
      }
      const bf16* tile = &s_ftile[buf][(size_t)(wave * 32) * H_];

      // prefetch next featsT region (one cacheline hint per lane)
      if (nt + 256 < N_)
        __builtin_prefetch(ftb + (size_t)(lane * 4) * N_ + n0 + 256, 0, 1);

#pragma unroll
      for (int sub = 0; sub < 2; ++sub) {
        const int nc = n0 + sub * 16;
        f32x8 c = {0.f, 0.f, 0.f, 0.f, 0.f, 0.f, 0.f, 0.f};
#pragma unroll
        for (int kc = 0; kc < 4; ++kc) {
          // B[k=h][n] from LDS tile: 32B contiguous per lane
          c = wmma_bf16(
              as[kc],
              load_contig16(tile + (size_t)(sub * 16 + colq) * H_ + kc * 32 + (hi << 4)),
              c);
        }
        const int ncol = nc + colq;
        const float f2v = f2b[ncol];
        float lg[8], mx = -3.0e38f;
#pragma unroll
        for (int r = 0; r < 8; ++r) {
          float d = (2.f * c[r] - s_s2[r + (hi << 3)] - f2v) * 10.f;  // /TAU
          lg[r] = d;
          mx = fmaxf(mx, d);
        }
        mx = fmaxf(mx, __shfl_xor(mx, 16, 32));  // combine both slot halves
        float se = 0.f;
#pragma unroll
        for (int r = 0; r < 8; ++r) {
          lg[r] = __expf(lg[r] - mx);
          se += lg[r];
        }
        se += __shfl_xor(se, 16, 32);
        float inv = 1.0f / se;
#pragma unroll
        for (int r = 0; r < 8; ++r) {
          float a = lg[r] * inv;
          apart[r] += a;
          s_stage[wave][r + (hi << 3)][sub * 16 + colq] = (bf16)a;
          if (iter == 2)
            out_attn[((size_t)(bidx * S_ + r + (hi << 3))) * N_ + ncol] = a;
        }
      }
      __syncthreads();  // LDS transpose visibility (uniform across block)

      // updates += attn(16x32) @ feats(32x128); B from featsT, 32B contiguous
      bf16x16 au = load_a_frag(&s_stage[wave][0][0], 32, lane);
#pragma unroll
      for (int ht = 0; ht < 8; ++ht) {
        cupd[ht] = wmma_bf16(
            au, load_contig16(ftb + (size_t)(ht * 16 + colq) * N_ + n0 + (hi << 4)),
            cupd[ht]);
      }
    }

    // ---- reduce denominators and numerators across waves ----
#pragma unroll
    for (int r = 0; r < 8; ++r) {
      float v = apart[r];
#pragma unroll
      for (int m = 1; m < 16; m <<= 1) v += __shfl_xor(v, m, 32);
      if (colq == 0) atomicAdd(&s_asum[r + (hi << 3)], v);
    }
#pragma unroll
    for (int ht = 0; ht < 8; ++ht)
#pragma unroll
      for (int r = 0; r < 8; ++r)
        atomicAdd(&s_upd[r + (hi << 3)][ht * 16 + colq], cupd[ht][r]);
    __syncthreads();

    // updates = num / (den + 1e-8)
    for (int i = tid; i < S_ * H_; i += 256) {
      int s = i >> 7;
      (&s_upd[0][0])[i] = (&s_upd[0][0])[i] / (s_asum[s] + 1e-8f);
    }
    __syncthreads();

    // ---- GRU cell (r,z,n order), 16x128 outputs, scalar dots ----
#pragma unroll
    for (int k = 0; k < 8; ++k) {
      int hh = jj + 16 * k;
      float gir = gru_bi[hh], giz = gru_bi[H_ + hh], gin = gru_bi[2 * H_ + hh];
      float ghr = gru_bh[hh], ghz = gru_bh[H_ + hh], ghn = gru_bh[2 * H_ + hh];
      const float* wir = gru_wi + (size_t)hh * H_;
      const float* wiz = gru_wi + (size_t)(H_ + hh) * H_;
      const float* win = gru_wi + (size_t)(2 * H_ + hh) * H_;
      const float* whr = gru_wh + (size_t)hh * H_;
      const float* whz = gru_wh + (size_t)(H_ + hh) * H_;
      const float* whn = gru_wh + (size_t)(2 * H_ + hh) * H_;
      for (int q = 0; q < H_; ++q) {
        float u = s_upd[sg][q], sl = s_slots[sg][q];
        gir += u * wir[q];
        giz += u * wiz[q];
        gin += u * win[q];
        ghr += sl * whr[q];
        ghz += sl * whz[q];
        ghn += sl * whn[q];
      }
      float r = 1.f / (1.f + __expf(-(gir + ghr)));
      float z = 1.f / (1.f + __expf(-(giz + ghz)));
      float nn = tanhf(gin + r * ghn);
      s_new[sg][hh] = (1.f - z) * nn + z * s_slots[sg][hh];
    }
    __syncthreads();

    // ---- LN(new) ----
    {
      float s = 0.f, sq = 0.f;
#pragma unroll
      for (int k = 0; k < 8; ++k) {
        float v = s_new[sg][jj + 16 * k];
        s += v;
        sq += v * v;
      }
#pragma unroll
      for (int m = 1; m < 16; m <<= 1) {
        s += __shfl_xor(s, m, 32);
        sq += __shfl_xor(sq, m, 32);
      }
      float mu = s * (1.0f / H_);
      float rstd = rsqrtf(sq * (1.0f / H_) - mu * mu + 1e-5f);
#pragma unroll
      for (int k = 0; k < 8; ++k) {
        int hh = jj + 16 * k;
        s_buf[sg][hh] = (s_new[sg][hh] - mu) * rstd * norm_g[hh] + norm_b[hh];
      }
    }
    __syncthreads();

    // ---- MLP layer 1 (GELU), into s_upd (reuse) ----
#pragma unroll
    for (int k = 0; k < 8; ++k) {
      int hh = jj + 16 * k;
      float acc = mlp_b1[hh];
      for (int q = 0; q < H_; ++q) acc += s_buf[sg][q] * mlp_w1[(size_t)q * H_ + hh];
      s_upd[sg][hh] = gelu_exact(acc);
    }
    __syncthreads();

    // ---- MLP layer 2 + residual -> slots ----
#pragma unroll
    for (int k = 0; k < 8; ++k) {
      int hh = jj + 16 * k;
      float acc = mlp_b2[hh];
      for (int q = 0; q < H_; ++q) acc += s_upd[sg][q] * mlp_w2[(size_t)q * H_ + hh];
      s_slots[sg][hh] = s_new[sg][hh] + 0.2f * acc;
    }
    __syncthreads();
  }

  for (int i = tid; i < S_ * H_; i += 256)
    out_slots[(size_t)bidx * S_ * H_ + i] = (&s_slots[0][0])[i];
}

extern "C" void kernel_launch(void* const* d_in, const int* in_sizes, int n_in,
                              void* d_out, int out_size, void* d_ws, size_t ws_size,
                              hipStream_t stream) {
  const float* x = (const float*)d_in[0];
  const float* w1 = (const float*)d_in[1];
  const float* b1 = (const float*)d_in[2];
  const float* g1 = (const float*)d_in[3];
  const float* bb1 = (const float*)d_in[4];
  const float* w2 = (const float*)d_in[5];
  const float* b2 = (const float*)d_in[6];
  const float* g2 = (const float*)d_in[7];
  const float* bb2 = (const float*)d_in[8];
  const float* slot_mu = (const float*)d_in[9];
  const float* gru_wi = (const float*)d_in[10];
  const float* gru_wh = (const float*)d_in[11];
  const float* gru_bi = (const float*)d_in[12];
  const float* gru_bh = (const float*)d_in[13];
  const float* mlp_w1 = (const float*)d_in[14];
  const float* mlp_b1 = (const float*)d_in[15];
  const float* mlp_w2 = (const float*)d_in[16];
  const float* mlp_b2 = (const float*)d_in[17];
  const float* norm_g = (const float*)d_in[18];
  const float* norm_b = (const float*)d_in[19];

  const size_t featsBytes = (size_t)B_ * N_ * H_ * sizeof(bf16);  // 64 MB
  bf16* feats = (bf16*)d_ws;
  bf16* featsT = (bf16*)((char*)d_ws + featsBytes);
  float* f2 = (float*)((char*)d_ws + 2 * featsBytes);

  float* out_slots = (float*)d_out;                    // [B,S,H]
  float* out_attn = out_slots + (size_t)B_ * S_ * H_;  // [B,S,N]

  feats_kernel<<<dim3(B_ * (N_ / 128)), dim3(256), 0, stream>>>(
      x, w1, b1, g1, bb1, w2, b2, g2, bb2, feats, featsT, f2);
  slots_kernel<<<dim3(B_), dim3(256), 0, stream>>>(
      feats, featsT, f2, slot_mu, gru_wi, gru_wh, gru_bi, gru_bh, mlp_w1,
      mlp_b1, mlp_w2, mlp_b2, norm_g, norm_b, out_slots, out_attn);
}
